// IPAGNNLayer_80994493268234
// MI455X (gfx1250) — matrix operand order repaired and checked by reading, not compiled, not run on hardware
//
#include <hip/hip_runtime.h>

#define BB 32
#define NN 2048
#define HH 128
#define MM (BB * NN)          // 65536 tokens
#define KK 256                // fused K = H(x) + H(h)
#define GOUT 512              // 4*H gate width
#define OUTW 513              // 4H + 1 (ip)

typedef __attribute__((ext_vector_type(16))) __bf16 v16bf;
typedef __attribute__((ext_vector_type(8)))  float  v8f;

__device__ __forceinline__ __bf16 f2bf(float f) {
    unsigned u = __builtin_bit_cast(unsigned, f);
    u += 0x7fffu + ((u >> 16) & 1u);                 // round-to-nearest-even
    unsigned short h = (unsigned short)(u >> 16);
    return __builtin_bit_cast(__bf16, h);
}

__device__ __forceinline__ float sigmoidf_fast(float x) {
    return 1.0f / (1.0f + __expf(-x));
}
__device__ __forceinline__ float tanhf_fast(float x) {
    return 1.0f - 2.0f / (__expf(2.0f * x) + 1.0f);
}

// ---------------------------------------------------------------------------
// Convert W = [Wx; Wh] (256 x 512 f32, row-major per source) into W^T bf16
// WT[n][k], n in [0,512), k in [0,256).
// ---------------------------------------------------------------------------
__global__ __launch_bounds__(256) void wconv_kernel(
    const float* __restrict__ Wx, const float* __restrict__ Wh,
    __bf16* __restrict__ WT)
{
    int idx = blockIdx.x * 256 + threadIdx.x;        // 512*256 elements, 512 blocks
    int n = idx >> 8;
    int k = idx & 255;
    float v = (k < HH) ? Wx[(size_t)k * GOUT + n]
                       : Wh[(size_t)(k - HH) * GOUT + n];
    WT[(size_t)n * KK + k] = f2bf(v);
}

// ---------------------------------------------------------------------------
// Fused LSTM cell: g = [X|Hp] @ W + b ; c' = sig(f)*c + sig(i)*tanh(g)
//                  h' = sig(o)*tanh(c')
// Block: 256 thr (8 waves). grid.x = M/128 row blocks, grid.y = 4 (32-ch blocks).
// Each wave: 16 rows x 32 channels across all 4 gates -> 8 wmma acc tiles.
// ---------------------------------------------------------------------------
__global__ __launch_bounds__(256) void lstm_cell_kernel(
    const float* __restrict__ X,  int xs,            // A, k in [0,128)
    const float* __restrict__ Hp, int hs,            // A, k in [128,256)
    const __bf16* __restrict__ WT,                   // [512][256] bf16
    const float* __restrict__ bias,                  // [512]
    const float* __restrict__ Cp, int cs,            // prev cell state
    float* __restrict__ Co, float* __restrict__ Ho, int os)
{
    __shared__ __align__(16) __bf16 sW[128 * KK];    // 64 KB W^T slab
    const int tid = threadIdx.x;
    const int chb = blockIdx.y * 32;                 // channel base (0..96)

    // Stage slab: local row r = gate*32 + j  <-  WT row gate*128 + chb + j
    {
        int r = tid >> 1, half = tid & 1;
        int g = r >> 5, j = r & 31;
        const uint4* src = (const uint4*)(WT + (size_t)(g * HH + chb + j) * KK + half * 128);
        uint4* dst = (uint4*)(sW + r * KK + half * 128);
#pragma unroll
        for (int i = 0; i < 16; ++i) dst[i] = src[i];
    }
    __syncthreads();

    const int wave = tid >> 5, lane = tid & 31;
    const int lrow = lane & 15, halfsel = lane >> 4;
    const int mbase = blockIdx.x * 128 + wave * 16;
    const int mrow = mbase + lrow;

    v8f acc[8] = {};                                  // [gate*2 + chtile]

#pragma unroll
    for (int ks = 0; ks < 8; ++ks) {
        const int k0 = ks * 32;
        // ---- A fragment: lane<16 -> K {k0..k0+7, k0+16..23}; lane>=16 -> +8
        const float* srcrow = (k0 < HH)
            ? (X  + (size_t)mrow * xs + k0 + halfsel * 8)
            : (Hp + (size_t)mrow * hs + (k0 - HH) + halfsel * 8);
        float4 c0a = ((const float4*)srcrow)[0];
        float4 c0b = ((const float4*)srcrow)[1];
        float4 c1a = ((const float4*)(srcrow + 16))[0];
        float4 c1b = ((const float4*)(srcrow + 16))[1];
        v16bf a;
        a[0]  = f2bf(c0a.x); a[1]  = f2bf(c0a.y); a[2]  = f2bf(c0a.z); a[3]  = f2bf(c0a.w);
        a[4]  = f2bf(c0b.x); a[5]  = f2bf(c0b.y); a[6]  = f2bf(c0b.z); a[7]  = f2bf(c0b.w);
        a[8]  = f2bf(c1a.x); a[9]  = f2bf(c1a.y); a[10] = f2bf(c1a.z); a[11] = f2bf(c1a.w);
        a[12] = f2bf(c1b.x); a[13] = f2bf(c1b.y); a[14] = f2bf(c1b.z); a[15] = f2bf(c1b.w);

        const int kb = k0 + halfsel * 8;              // B: lanes 0-15 K {0..7,16..23}, 16-31 +8
#pragma unroll
        for (int nt = 0; nt < 8; ++nt) {
            int nl = (nt >> 1) * 32 + (nt & 1) * 16 + lrow;
            struct { uint4 lo, hi; } bq = { *(const uint4*)(sW + nl * KK + kb), *(const uint4*)(sW + nl * KK + kb + 16) };
            v16bf bf = __builtin_bit_cast(v16bf, bq);
            acc[nt] = __builtin_amdgcn_wmma_f32_16x16x32_bf16(
                false, a, false, bf, (short)0, acc[nt], false, false);
        }
    }

    // ---- fused gate epilogue (C tile: vgpr r, col=lrow, row=r+8*halfsel)
#pragma unroll
    for (int t = 0; t < 2; ++t) {
        const int ch = chb + t * 16 + lrow;           // channel in [0,128)
        const float bi = bias[0 * HH + ch];
        const float bf_ = bias[1 * HH + ch];
        const float bg = bias[2 * HH + ch];
        const float bo = bias[3 * HH + ch];
#pragma unroll
        for (int r = 0; r < 8; ++r) {
            const int row = mbase + r + 8 * halfsel;
            float gi = acc[0 + t][r] + bi;
            float gf = acc[2 + t][r] + bf_;
            float gg = acc[4 + t][r] + bg;
            float go = acc[6 + t][r] + bo;
            float cp = Cp[(size_t)row * cs + ch];
            float cn = sigmoidf_fast(gf) * cp + sigmoidf_fast(gi) * tanhf_fast(gg);
            float hn = sigmoidf_fast(go) * tanhf_fast(cn);
            Co[(size_t)row * os + ch] = cn;
            Ho[(size_t)row * os + ch] = hn;
        }
    }
}

// ---------------------------------------------------------------------------
// Overwrite exit/raise node rows of E with the OLD state (raise_in_ipagnn mask)
// grid = B*2 blocks x 128 threads
// ---------------------------------------------------------------------------
__global__ void mask_kernel(float* __restrict__ E,
    const float* __restrict__ c0, const float* __restrict__ h0,
    const float* __restrict__ c1, const float* __restrict__ h1,
    const int* __restrict__ exit_idx, const int* __restrict__ raise_idx)
{
    int b = blockIdx.x >> 1;
    int n = (blockIdx.x & 1) ? raise_idx[b] : exit_idx[b];
    size_t node = (size_t)b * NN + n;
    int t = threadIdx.x;                              // 0..127
    float* dst = E + node * GOUT;
    dst[0 * HH + t] = c0[node * HH + t];
    dst[1 * HH + t] = h0[node * HH + t];
    dst[2 * HH + t] = c1[node * HH + t];
    dst[3 * HH + t] = h1[node * HH + t];
}

// ---------------------------------------------------------------------------
// Raise/branch decisions: wave-per-node 4 dots over 512 ch, 2-way softmaxes.
// Outputs wr/wt/wf = probability * instruction_pointer.
// ---------------------------------------------------------------------------
__global__ __launch_bounds__(256) void dec_kernel(
    const float* __restrict__ E,
    const float* __restrict__ rW, const float* __restrict__ rb,
    const float* __restrict__ bW, const float* __restrict__ bb,
    const float* __restrict__ ip, const int* __restrict__ exit_idx,
    float* __restrict__ wr, float* __restrict__ wt, float* __restrict__ wf)
{
    int wave = threadIdx.x >> 5, lane = threadIdx.x & 31;
    int node = blockIdx.x * 8 + wave;
    const float* e = E + (size_t)node * GOUT;
    float r0 = 0.f, r1 = 0.f, d0 = 0.f, d1 = 0.f;
#pragma unroll
    for (int it = 0; it < 16; ++it) {
        int ch = lane + it * 32;
        float v = e[ch];
        float2 rw = ((const float2*)rW)[ch];
        float2 bw = ((const float2*)bW)[ch];
        r0 += v * rw.x; r1 += v * rw.y;
        d0 += v * bw.x; d1 += v * bw.y;
    }
#pragma unroll
    for (int off = 16; off; off >>= 1) {
        r0 += __shfl_xor(r0, off, 32);
        r1 += __shfl_xor(r1, off, 32);
        d0 += __shfl_xor(d0, off, 32);
        d1 += __shfl_xor(d1, off, 32);
    }
    if (lane == 0) {
        r0 += rb[0]; r1 += rb[1]; d0 += bb[0]; d1 += bb[1];
        float p_raise = 1.f / (1.f + __expf(r1 - r0));   // softmax_0 = sig(r0-r1)
        int b = node >> 11, n = node & (NN - 1);
        if (n == exit_idx[b]) p_raise = 0.f;             // raise_dec := [0,1] at exit
        float p_nr = 1.f - p_raise;
        float pt = p_nr * (1.f / (1.f + __expf(d1 - d0)));
        float pf = p_nr - pt;
        float ipv = ip[node];
        wr[node] = p_raise * ipv;
        wt[node] = pt * ipv;
        wf[node] = pf * ipv;
    }
}

// ---------------------------------------------------------------------------
// Soft-edge scatter-add (segment_sum) into out[B,N,513] via f32 atomics.
// block-per-node; channels + ip slot.
// ---------------------------------------------------------------------------
__global__ __launch_bounds__(256) void scatter_kernel(
    const float* __restrict__ E,
    const float* __restrict__ wr, const float* __restrict__ wt,
    const float* __restrict__ wf,
    const int* __restrict__ ri, const int* __restrict__ ti,
    const int* __restrict__ fi,
    float* __restrict__ out)
{
    int node = blockIdx.x;
    int b = node >> 11;
    size_t base = (size_t)b * NN;
    float wwr = wr[node], wwt = wt[node], wwf = wf[node];
    size_t tr = (base + ri[node]) * (size_t)OUTW;
    size_t tt = (base + ti[node]) * (size_t)OUTW;
    size_t tf = (base + fi[node]) * (size_t)OUTW;
    const float* e = E + (size_t)node * GOUT;
#pragma unroll
    for (int it = 0; it < 2; ++it) {
        int ch = threadIdx.x + it * 256;
        float v = e[ch];
        atomicAdd(out + tr + ch, v * wwr);
        atomicAdd(out + tt + ch, v * wwt);
        atomicAdd(out + tf + ch, v * wwf);
    }
    if (threadIdx.x == 0) {
        atomicAdd(out + tr + 512, wwr);
        atomicAdd(out + tt + 512, wwt);
        atomicAdd(out + tf + 512, wwf);
    }
}

// ---------------------------------------------------------------------------
// Finalize: divide by (ip_new + 1e-7) when live, else restore old state.
// ---------------------------------------------------------------------------
__global__ __launch_bounds__(256) void final_kernel(
    float* __restrict__ out,
    const float* __restrict__ c0, const float* __restrict__ h0,
    const float* __restrict__ c1, const float* __restrict__ h1,
    const float* __restrict__ ip,
    const int* __restrict__ cstep, const int* __restrict__ slim)
{
    int node = blockIdx.x;
    int b = node >> 11;
    bool live = cstep[b] < slim[b];
    float* o = out + (size_t)node * OUTW;
    if (live) {
        float denom = o[512] + 1e-7f;
#pragma unroll
        for (int it = 0; it < 2; ++it) {
            int ch = threadIdx.x + it * 256;
            o[ch] = o[ch] / denom;
        }
    } else {
#pragma unroll
        for (int it = 0; it < 2; ++it) {
            int ch = threadIdx.x + it * 256;
            int part = ch >> 7, off = ch & (HH - 1);
            const float* src = (part == 0) ? c0 : (part == 1) ? h0
                             : (part == 2) ? c1 : h1;
            o[ch] = src[(size_t)node * HH + off];
        }
        if (threadIdx.x == 0) o[512] = ip[node];
    }
}

// ---------------------------------------------------------------------------
extern "C" void kernel_launch(void* const* d_in, const int* in_sizes, int n_in,
                              void* d_out, int out_size, void* d_ws, size_t ws_size,
                              hipStream_t stream) {
    const float* x   = (const float*)d_in[0];
    const float* c0  = (const float*)d_in[1];
    const float* h0  = (const float*)d_in[2];
    const float* c1  = (const float*)d_in[3];
    const float* h1  = (const float*)d_in[4];
    const float* ip  = (const float*)d_in[5];
    const float* Wx0 = (const float*)d_in[6];
    const float* Wh0 = (const float*)d_in[7];
    const float* b0  = (const float*)d_in[8];
    const float* Wx1 = (const float*)d_in[9];
    const float* Wh1 = (const float*)d_in[10];
    const float* b1  = (const float*)d_in[11];
    const float* rW  = (const float*)d_in[12];
    const float* rb  = (const float*)d_in[13];
    const float* bW  = (const float*)d_in[14];
    const float* bb  = (const float*)d_in[15];
    const int* ti    = (const int*)d_in[16];
    const int* fi    = (const int*)d_in[17];
    const int* ri    = (const int*)d_in[18];
    const int* ei    = (const int*)d_in[19];
    const int* rni   = (const int*)d_in[20];
    const int* cs    = (const int*)d_in[21];
    const int* sl    = (const int*)d_in[22];
    float* out = (float*)d_out;

    char* ws = (char*)d_ws;
    __bf16* WT0 = (__bf16*)ws;                             // 512*256 bf16
    __bf16* WT1 = WT0 + (size_t)GOUT * KK;
    float* E  = (float*)(ws + 2 * (size_t)GOUT * KK * sizeof(__bf16));
    float* wr = E + (size_t)MM * GOUT;
    float* wt = wr + MM;
    float* wf = wt + MM;

    hipMemsetAsync(d_out, 0, (size_t)out_size * sizeof(float), stream);

    wconv_kernel<<<512, 256, 0, stream>>>(Wx0, Wh0, WT0);
    wconv_kernel<<<512, 256, 0, stream>>>(Wx1, Wh1, WT1);

    dim3 g(MM / 128, 4);
    // cell 0: writes c0n -> E[:,0:128], h0n -> E[:,128:256]
    lstm_cell_kernel<<<g, 256, 0, stream>>>(x, HH, h0, HH, WT0, b0, c0, HH,
                                            E + 0, E + 128, GOUT);
    // cell 1: input = h0n (E slice, stride 512); writes c1n/h1n -> E[:,256:512]
    lstm_cell_kernel<<<g, 256, 0, stream>>>(E + 128, GOUT, h1, HH, WT1, b1, c1, HH,
                                            E + 256, E + 384, GOUT);

    mask_kernel<<<BB * 2, 128, 0, stream>>>(E, c0, h0, c1, h1, ei, rni);
    dec_kernel<<<MM / 8, 256, 0, stream>>>(E, rW, rb, bW, bb, ip, ei, wr, wt, wf);
    scatter_kernel<<<MM, 256, 0, stream>>>(E, wr, wt, wf, ri, ti, fi, out);
    final_kernel<<<MM, 256, 0, stream>>>(out, c0, h0, c1, h1, ip, cs, sl);
}
